// ELTE_27152783245375
// MI455X (gfx1250) — matrix operand order repaired and checked by reading, not compiled, not run on hardware
//
#include <hip/hip_runtime.h>
#include <stdint.h>

typedef __attribute__((ext_vector_type(16))) _Float16 v16h;
typedef __attribute__((ext_vector_type(8)))  float    v8f;
typedef __attribute__((ext_vector_type(4)))  float    vf4;

#define HH   128
#define WW   128
#define HWP  (HH*WW)
#define CIN  64
#define HID  256
#define HIDH 128
#define NQ   65536
#define PITCH 264   // 256 halves + 8 pad (528B rows, 16B aligned)

// ---------------------------------------------------------------------------
// Prep: convert MLP weights to f16; zero-pad W2 (3x256) to 16x256.
// ---------------------------------------------------------------------------
__global__ void ELTE_prep_w(const float* __restrict__ w0, const float* __restrict__ w1,
                            const float* __restrict__ w2,
                            _Float16* __restrict__ w0h, _Float16* __restrict__ w1h,
                            _Float16* __restrict__ w2h) {
  const int stride = gridDim.x * blockDim.x;
  int i = blockIdx.x * blockDim.x + threadIdx.x;
  for (int k = i; k < HID * HID; k += stride) {
    w0h[k] = (_Float16)w0[k];
    w1h[k] = (_Float16)w1[k];
  }
  for (int k = i; k < 16 * HID; k += stride) {
    float v = (k < 3 * HID) ? w2[k] : 0.0f;
    w2h[k] = (_Float16)v;
  }
}

// ---------------------------------------------------------------------------
// 3x3 SAME conv, fp32, output transposed to channel-last: outT[pixel][OC].
// One block = 8x8 pixel tile, blockDim.x == OC, input patch staged in LDS.
// ---------------------------------------------------------------------------
__global__ __launch_bounds__(256) void ELTE_conv3x3_t(
    const float* __restrict__ feat, const float* __restrict__ w,
    const float* __restrict__ bias, float* __restrict__ outT, int OC) {
  __shared__ float patch[CIN * 100];  // 64 ch x 10x10 halo tile (25.6 KB)
  const int tileId = blockIdx.x;      // 16x16 tiles of 8x8
  const int ty0 = (tileId >> 4) * 8 - 1;
  const int tx0 = (tileId & 15) * 8 - 1;
  for (int idx = threadIdx.x; idx < CIN * 100; idx += blockDim.x) {
    int ic = idx / 100, r = idx % 100;
    int ly = r / 10, lx = r % 10;
    int gy = ty0 + ly, gx = tx0 + lx;
    float v = 0.0f;
    if (gy >= 0 && gy < HH && gx >= 0 && gx < WW)
      v = feat[ic * HWP + gy * WW + gx];
    patch[idx] = v;
  }
  __syncthreads();
  const int oc = threadIdx.x;
  const float bv = bias[oc];
  const float* wr = w + oc * (CIN * 9);
  for (int pc = 0; pc < 4; ++pc) {  // 4 chunks of 16 pixels
    float acc[16];
#pragma unroll
    for (int p = 0; p < 16; ++p) acc[p] = bv;
    for (int ic = 0; ic < CIN; ++ic) {
      const float* pb = patch + ic * 100;
      const float* wb = wr + ic * 9;
#pragma unroll
      for (int dy = 0; dy < 3; ++dy) {
#pragma unroll
        for (int dx = 0; dx < 3; ++dx) {
          const float wv = wb[dy * 3 + dx];
          const float* pr = pb + dy * 10 + dx;
#pragma unroll
          for (int p = 0; p < 16; ++p) {
            int pix = pc * 16 + p;
            acc[p] = fmaf(wv, pr[(pix >> 3) * 10 + (pix & 7)], acc[p]);
          }
        }
      }
    }
#pragma unroll
    for (int p = 0; p < 16; ++p) {
      int pix = pc * 16 + p;
      int gy = ty0 + 1 + (pix >> 3), gx = tx0 + 1 + (pix & 7);
      outT[(gy * WW + gx) * OC + oc] = acc[p];
    }
  }
}

// ---------------------------------------------------------------------------
// Fused LTE query kernel. One wave = 16 rows = 4 queries x 4 corners.
// GEMM convention: A = weight tile (16 out x 32 k) from LDS, B = activations
// (32 k x 16 rows) in registers. D (column-striped) -> next-layer B fragment
// via half-wave shuffle + f16 pack, no LDS staging of activations.
// ---------------------------------------------------------------------------
__device__ __forceinline__ v8f ELTE_mlp_tile(const _Float16* Wlds, const v16h* B,
                                             int mtile, int hi, int l15,
                                             const float* __restrict__ bias) {
  union { v8f v; vf4 q[2]; float f[8]; } c;
  const vf4* bp = (const vf4*)(bias + mtile * 16 + hi * 8);
  c.q[0] = bp[0];
  c.q[1] = bp[1];
  const _Float16* wrow = Wlds + (mtile * 16 + l15) * PITCH;
#pragma unroll
  for (int s = 0; s < 8; ++s) {
    union { v16h v; vf4 q[2]; } A;
    const int off = 32 * s + 16 * hi;  // A frag: k {off..off+15}
    A.q[0] = *(const vf4*)(wrow + off);
    A.q[1] = *(const vf4*)(wrow + off + 8);
    c.v = __builtin_amdgcn_wmma_f32_16x16x32_f16(false, A.v, false, B[s],
                                                 (short)0, c.v, false, false);
  }
  return c.v;
}

__global__ __launch_bounds__(256) void ELTE_lte_query(
    const float* __restrict__ inp, const float* __restrict__ coord,
    const float* __restrict__ cell,
    const float* __restrict__ coefT, const float* __restrict__ freqT,
    const float* __restrict__ cfw, const float* __restrict__ pwm,
    const _Float16* __restrict__ w0h, const _Float16* __restrict__ w1h,
    const _Float16* __restrict__ w2h,
    const float* __restrict__ b0, const float* __restrict__ b1,
    const float* __restrict__ b2, float* __restrict__ out) {
  __shared__ _Float16 sW0[HID * PITCH];  // 132 KB
  __shared__ _Float16 sW1[HID * PITCH];  // 132 KB
  __shared__ _Float16 sW2[16 * PITCH];   // 8.25 KB

  // Stage f16 weights into padded LDS (uint = 2 halves).
  for (int i = threadIdx.x; i < HID * (HID / 2); i += blockDim.x) {
    int r = i >> 7, c = i & 127;
    *(uint32_t*)&sW0[r * PITCH + 2 * c] = ((const uint32_t*)w0h)[i];
    *(uint32_t*)&sW1[r * PITCH + 2 * c] = ((const uint32_t*)w1h)[i];
  }
  for (int i = threadIdx.x; i < 16 * (HID / 2); i += blockDim.x) {
    int r = i >> 7, c = i & 127;
    *(uint32_t*)&sW2[r * PITCH + 2 * c] = ((const uint32_t*)w2h)[i];
  }
  __syncthreads();

  const int lane = threadIdx.x & 31;
  const int hi = lane >> 4;   // lane group: 0 -> k low half, 1 -> k high half
  const int l15 = lane & 15;  // row within 16-row tile
  const int wavesPerBlk = blockDim.x >> 5;
  const int waveId = threadIdx.x >> 5;
  const int nWaves = wavesPerBlk * gridDim.x;
  const float PI = 3.14159265358979323846f;

#pragma clang loop unroll(disable)
  for (int tile = blockIdx.x * wavesPerBlk + waveId; tile < NQ / 4; tile += nWaves) {
    // Opaque memory clobber: forbids LICM from hoisting the (loop-invariant)
    // LDS weight-fragment / bias / table loads out of the tile loop, which
    // would otherwise spill >700 VGPRs to scratch. Weights must be re-read
    // from LDS every iteration -- that is the design.
    asm volatile("" ::: "memory");

    const int qbase = tile * 4;
    const int q = qbase + (l15 >> 2);
    const int crn = l15 & 3;  // corner k: (vx,vy) = (-- , -+ , +- , ++)

    // ---- per-row corner geometry --------------------------------------
    const float cy = coord[2 * q], cx = coord[2 * q + 1];
    const float vx = (crn & 2) ? 1.0f : -1.0f;
    const float vy = (crn & 1) ? 1.0f : -1.0f;
    const float rr = 1.0f / 128.0f;
    float cs0 = fminf(fmaxf(cy + vx * rr + 1e-6f, -1.0f + 1e-6f), 1.0f - 1e-6f);
    float cs1 = fminf(fmaxf(cx + vy * rr + 1e-6f, -1.0f + 1e-6f), 1.0f - 1e-6f);
    int iy = (int)rintf((cs0 + 1.0f) * 64.0f - 0.5f);
    int ix = (int)rintf((cs1 + 1.0f) * 64.0f - 0.5f);
    iy = min(max(iy, 0), HH - 1);
    ix = min(max(ix, 0), WW - 1);
    const int flat = iy * WW + ix;
    const float rel0 = (cy - ((2.0f * (float)iy + 1.0f) / 128.0f - 1.0f)) * 128.0f;
    const float rel1 = (cx - ((2.0f * (float)ix + 1.0f) / 128.0f - 1.0f)) * 128.0f;
    const float rc0 = cell[2 * q] * 128.0f;
    const float rc1 = cell[2 * q + 1] * 128.0f;

    // ---- build x = q_coef * [cos(pi f); sin(pi f)] as B fragments -----
    // lane hi=0 holds k = 32s + 0..15 of its row; hi=1 holds k = 32s + 16..31
    v16h Bx[8];
    const float* fr = freqT + (size_t)flat * HIDH;
    const float* cf = coefT + (size_t)flat * HID;
#pragma unroll
    for (int t = 0; t < 4; ++t) {
      const int f0 = t * 32 + hi * 16;
      float m[16];
#pragma unroll
      for (int j = 0; j < 16; ++j) {
        const int f = f0 + j;
        const float fv = fr[f];
        const float mod = fv * (rel0 * cfw[2 * f] + rel1 * cfw[2 * f + 1]) +
                          (rc0 * pwm[2 * f] + rc1 * pwm[2 * f + 1]);
        m[j] = PI * mod;
      }
      union { v16h v; _Float16 h[16]; } lo, hs;
#pragma unroll
      for (int j = 0; j < 16; ++j) {
        lo.h[j] = (_Float16)(cf[f0 + j] * __cosf(m[j]));         // ch < 128: cos
        hs.h[j] = (_Float16)(cf[128 + f0 + j] * __sinf(m[j]));   // ch >=128: sin
      }
      Bx[t] = lo.v;
      Bx[t + 4] = hs.v;
    }

    // ---- layer 0: h0 = relu(W0 x + b0) -> B fragments -----------------
    v16h B1[8];
#pragma unroll
    for (int sp = 0; sp < 8; ++sp) {
      v8f a = ELTE_mlp_tile(sW0, Bx, 2 * sp, hi, l15, b0);
      v8f b = ELTE_mlp_tile(sW0, Bx, 2 * sp + 1, hi, l15, b0);
      union { v16h v; _Float16 h[16]; } u;
#pragma unroll
      for (int j = 0; j < 8; ++j) {
        float av = fmaxf(a[j], 0.0f);
        float bv = fmaxf(b[j], 0.0f);
        float pa = __shfl_xor(av, 16);  // partner half-wave
        float pb = __shfl_xor(bv, 16);
        float loV = hi ? pb : av;  // k halves 0..7  (outs 0..7 of its tile)
        float hiV = hi ? bv : pa;  // k halves 8..15 (outs 8..15 of its tile)
        u.h[j] = (_Float16)loV;
        u.h[8 + j] = (_Float16)hiV;
      }
      B1[sp] = u.v;
    }

    // ---- layer 1: h1 = relu(W1 h0 + b1) -> B fragments ----------------
    v16h B2[8];
#pragma unroll
    for (int sp = 0; sp < 8; ++sp) {
      v8f a = ELTE_mlp_tile(sW1, B1, 2 * sp, hi, l15, b1);
      v8f b = ELTE_mlp_tile(sW1, B1, 2 * sp + 1, hi, l15, b1);
      union { v16h v; _Float16 h[16]; } u;
#pragma unroll
      for (int j = 0; j < 8; ++j) {
        float av = fmaxf(a[j], 0.0f);
        float bv = fmaxf(b[j], 0.0f);
        float pa = __shfl_xor(av, 16);
        float pb = __shfl_xor(bv, 16);
        u.h[j] = (_Float16)(hi ? pb : av);
        u.h[8 + j] = (_Float16)(hi ? bv : pa);
      }
      B2[sp] = u.v;
    }

    // ---- layer 2: pred = W2pad h1 + b2 (N padded to 16, rows 3..15 = 0)
    union { v8f v; vf4 qv[2]; float f[8]; } c2;
#pragma unroll
    for (int j = 0; j < 8; ++j)
      c2.f[j] = (hi == 0 && j < 3) ? b2[j] : 0.0f;
    const _Float16* w2row = sW2 + l15 * PITCH;
#pragma unroll
    for (int s = 0; s < 8; ++s) {
      union { v16h v; vf4 qv[2]; } A;
      const int off = 32 * s + 16 * hi;
      A.qv[0] = *(const vf4*)(w2row + off);
      A.qv[1] = *(const vf4*)(w2row + off + 8);
      c2.v = __builtin_amdgcn_wmma_f32_16x16x32_f16(false, A.v, false, B2[s],
                                                    (short)0, c2.v, false, false);
    }
    // lanes 0..15: lane = row, c2.f[0..2] = pred channels.

    // ---- area weights (wts = area[::-1]/tot, 3-k == k^3) --------------
    const float area = fabsf(rel0 * rel1) + 1e-9f;
    const float arev = __shfl_xor(area, 3);
    float ts = area + __shfl_xor(area, 1);
    const float tot = ts + __shfl_xor(ts, 2);
    const float wgt = arev / tot;
    float p0 = c2.f[0] * wgt, p1 = c2.f[1] * wgt, p2 = c2.f[2] * wgt;
    p0 += __shfl_xor(p0, 1); p0 += __shfl_xor(p0, 2);
    p1 += __shfl_xor(p1, 1); p1 += __shfl_xor(p1, 2);
    p2 += __shfl_xor(p2, 1); p2 += __shfl_xor(p2, 2);

    if (lane < 16 && (lane & 3) == 0) {
      // bilinear border sample of inp at (cy,cx), align_corners=False
      float y = fminf(fmaxf((cy + 1.0f) * 64.0f - 0.5f, 0.0f), 127.0f);
      float x = fminf(fmaxf((cx + 1.0f) * 64.0f - 0.5f, 0.0f), 127.0f);
      float y0f = floorf(y), x0f = floorf(x);
      float wy = y - y0f, wx = x - x0f;
      int y0 = (int)y0f, x0 = (int)x0f;
      int y1 = min(y0 + 1, HH - 1), x1 = min(x0 + 1, WW - 1);
      float p[3] = {p0, p1, p2};
#pragma unroll
      for (int ch = 0; ch < 3; ++ch) {
        const float* ip = inp + ch * HWP;
        float v00 = ip[y0 * WW + x0], v01 = ip[y0 * WW + x1];
        float v10 = ip[y1 * WW + x0], v11 = ip[y1 * WW + x1];
        float bil = v00 * (1.0f - wy) * (1.0f - wx) + v01 * (1.0f - wy) * wx +
                    v10 * wy * (1.0f - wx) + v11 * wy * wx;
        out[q * 3 + ch] = p[ch] + bil;
      }
    }
  }
}

// ---------------------------------------------------------------------------
extern "C" void kernel_launch(void* const* d_in, const int* in_sizes, int n_in,
                              void* d_out, int out_size, void* d_ws, size_t ws_size,
                              hipStream_t stream) {
  (void)in_sizes; (void)n_in; (void)out_size; (void)ws_size;
  const float* inp    = (const float*)d_in[0];
  const float* feat   = (const float*)d_in[1];
  const float* coord  = (const float*)d_in[2];
  const float* cell   = (const float*)d_in[3];
  const float* coef_w = (const float*)d_in[4];
  const float* coef_b = (const float*)d_in[5];
  const float* freq_w = (const float*)d_in[6];
  const float* freq_b = (const float*)d_in[7];
  const float* cfw    = (const float*)d_in[8];   // coord_fc_w [128][2]
  const float* pwm    = (const float*)d_in[9];   // phase_w    [128][2]
  const float* w0     = (const float*)d_in[10];
  const float* b0     = (const float*)d_in[11];
  const float* w1     = (const float*)d_in[12];
  const float* b1     = (const float*)d_in[13];
  const float* w2     = (const float*)d_in[14];
  const float* b2     = (const float*)d_in[15];
  float* out = (float*)d_out;

  char* ws = (char*)d_ws;
  float* coefT = (float*)ws;                                   // 16384*256 f32 = 16 MB
  float* freqT = (float*)(ws + (size_t)HWP * HID * 4);         // 16384*128 f32 =  8 MB
  _Float16* w0h = (_Float16*)(ws + (size_t)HWP * HID * 4 + (size_t)HWP * HIDH * 4);
  _Float16* w1h = w0h + HID * HID;
  _Float16* w2h = w1h + HID * HID;

  ELTE_prep_w<<<64, 256, 0, stream>>>(w0, w1, w2, w0h, w1h, w2h);
  ELTE_conv3x3_t<<<256, 256, 0, stream>>>(feat, coef_w, coef_b, coefT, HID);
  ELTE_conv3x3_t<<<256, 128, 0, stream>>>(feat, freq_w, freq_b, freqT, HIDH);
  ELTE_lte_query<<<512, 256, 0, stream>>>(inp, coord, cell, coefT, freqT,
                                          cfw, pwm, w0h, w1h, w2h,
                                          b0, b1, b2, out);
}
